// MegalodonAttention_90142773609262
// MI455X (gfx1250) — matrix operand order repaired and checked by reading, not compiled, not run on hardware
//
#include <hip/hip_runtime.h>
#include <hip/hip_bf16.h>
#include <math.h>

typedef __attribute__((ext_vector_type(16))) _Float16 v16h;
typedef __attribute__((ext_vector_type(8)))  _Float16 v8h;
typedef __attribute__((ext_vector_type(8)))  float    v8f;

#define BATCH   4
#define SEQ     4096
#define HEADS   16
#define DHEAD   128
#define DVAL    128
#define CHUNKL  1024
#define QTILE   128     // q rows per workgroup (8 waves x 16 rows)
#define KTILE   32      // keys per inner block
#define NWAVE   8

// -ln(10000)/64  (RoPE half = 64)
#define NEG_LOGB_OVER_HALF (-0.14391156831212787f)
#define QK_SCALE           (0.08838834764831845f)   // 1/sqrt(128)

__device__ __forceinline__ float rowmax16(float v) {
  v = fmaxf(v, __shfl_xor(v, 1));
  v = fmaxf(v, __shfl_xor(v, 2));
  v = fmaxf(v, __shfl_xor(v, 4));
  v = fmaxf(v, __shfl_xor(v, 8));
  return v;
}
__device__ __forceinline__ float rowsum16(float v) {
  v += __shfl_xor(v, 1);
  v += __shfl_xor(v, 2);
  v += __shfl_xor(v, 4);
  v += __shfl_xor(v, 8);
  return v;
}

__global__ __launch_bounds__(256)
void mega_attn_kernel(const float* __restrict__ q,
                      const float* __restrict__ k,
                      const float* __restrict__ v,
                      const int*   __restrict__ start_index_p,
                      float* __restrict__ out)
{
  // K tile, post-RoPE f16, row-major [key][dh]      : 32*128*2 = 8 KB
  // V tile, f16, transposed [dv][key]               : 128*32*2 = 8 KB
  // P scratch per wave, [qrow][key]                 : 8*16*32*2 = 8 KB
  __shared__ _Float16 ldsK[KTILE * DHEAD];
  __shared__ _Float16 ldsVT[DVAL * KTILE];
  __shared__ _Float16 ldsP[NWAVE][16 * KTILE];

  const int qt = blockIdx.x;       // query tile within chunk: 0..7
  const int ch = blockIdx.y;       // flattened (b * nc + c): 0..15
  const int h  = blockIdx.z;       // head
  const int b  = ch >> 2;
  const int c  = ch & 3;
  const int bt = b * SEQ + c * CHUNKL;   // global row of chunk start

  const int tid  = threadIdx.x;
  const int wave = tid >> 5;
  const int lane = tid & 31;
  const int ln16 = lane & 15;
  const int lh   = lane >> 4;      // which 16-lane half (A/B layout group)

  const float start = (float)start_index_p[0];

  // -------- load Q tile for this wave into WMMA A-fragments, apply RoPE ----
  // A-layout (16-bit A 16x32): lane holds row = lane%16; half g = lane/16
  // holds k-offsets {8g..8g+7} (regs 0-3) and {16+8g..16+8g+7} (regs 4-7).
  // Across the 4 dh-steps of 32, this lane owns dh = 8*lh + 16*m + j, m=0..7.
  // RoPE pairs dh <-> dh+64, i.e. run m <-> m+4: stays inside the lane.
  const int wq   = qt * QTILE + wave * 16;      // wave's first q row (chunk-local)
  const int qrow = wq + ln16;                   // this lane's q row
  const size_t qoff = ((size_t)(bt + qrow) * HEADS + h) * DHEAD;

  float qf[64];
  #pragma unroll
  for (int m = 0; m < 8; ++m) {
    const float4 x0 = *(const float4*)(q + qoff + 16 * m + 8 * lh);
    const float4 x1 = *(const float4*)(q + qoff + 16 * m + 8 * lh + 4);
    qf[m*8+0]=x0.x; qf[m*8+1]=x0.y; qf[m*8+2]=x0.z; qf[m*8+3]=x0.w;
    qf[m*8+4]=x1.x; qf[m*8+5]=x1.y; qf[m*8+6]=x1.z; qf[m*8+7]=x1.w;
  }
  {
    const float pos = start + (float)qrow;
    #pragma unroll
    for (int m = 0; m < 4; ++m) {
      #pragma unroll
      for (int j = 0; j < 8; ++j) {
        const int i = 8 * lh + 16 * m + j;           // i < 64
        const float ang = pos * __expf((float)i * NEG_LOGB_OVER_HALF);
        float sn, cs;
        __sincosf(ang, &sn, &cs);
        const float a = qf[m*8+j];
        const float bq = qf[(m+4)*8+j];
        qf[m*8+j]     = a * cs - bq * sn;
        qf[(m+4)*8+j] = bq * cs + a * sn;
      }
    }
  }
  v16h aq[4];
  #pragma unroll
  for (int s = 0; s < 4; ++s) {
    #pragma unroll
    for (int j = 0; j < 8; ++j) {
      aq[s][j]     = (_Float16)(qf[(2*s)*8   + j] * QK_SCALE);
      aq[s][j + 8] = (_Float16)(qf[(2*s+1)*8 + j] * QK_SCALE);
    }
  }

  // -------- flash state ----------------------------------------------------
  v8f oacc[8];                     // out tile 16 x 128 (8 dv-subtiles), f32
  #pragma unroll
  for (int d = 0; d < 8; ++d)
    #pragma unroll
    for (int e = 0; e < 8; ++e) oacc[d][e] = 0.0f;
  float mrow[8], lrow[8];
  #pragma unroll
  for (int r = 0; r < 8; ++r) { mrow[r] = -INFINITY; lrow[r] = 0.0f; }

  // staging roles (256 threads): 8 threads per key row
  const int stg_key = tid >> 3;    // 0..31
  const int stg_seg = tid & 7;     // 0..7

  // RoPE inverse frequencies for this thread's staging lanes (loop-invariant)
  float kfreq[8];
  #pragma unroll
  for (int j = 0; j < 8; ++j)
    kfreq[j] = __expf((float)(stg_seg * 8 + j) * NEG_LOGB_OVER_HALF);

  const int nkb = 4 * (qt + 1);    // key blocks needed by this q tile

  // ---- software pipeline: raw f32 K/V for the block being transformed -----
  float4 ka0, ka1, kb0, kb1, va0, va1, va2, va3;
  {
    const int tk = stg_key;                               // block 0
    const float* kp = k + ((size_t)(bt + tk) * HEADS + h) * DHEAD + stg_seg * 8;
    ka0 = *(const float4*)(kp);      ka1 = *(const float4*)(kp + 4);
    kb0 = *(const float4*)(kp + 64); kb1 = *(const float4*)(kp + 68);
    const float* vp = v + ((size_t)(bt + tk) * HEADS + h) * DVAL + stg_seg * 16;
    va0 = *(const float4*)(vp);      va1 = *(const float4*)(vp + 4);
    va2 = *(const float4*)(vp + 8);  va3 = *(const float4*)(vp + 12);
  }

  for (int kb = 0; kb < nkb; ++kb) {
    const int kbase = kb * KTILE;

    // ---- transform staged registers -> LDS (RoPE K, transpose V) ----------
    {
      const float posk = start + (float)(kbase + stg_key);
      const float k1v[8] = {ka0.x,ka0.y,ka0.z,ka0.w,ka1.x,ka1.y,ka1.z,ka1.w};
      const float k2v[8] = {kb0.x,kb0.y,kb0.z,kb0.w,kb1.x,kb1.y,kb1.z,kb1.w};
      v8h o1, o2;
      #pragma unroll
      for (int j = 0; j < 8; ++j) {
        const float ang = posk * kfreq[j];
        float sn, cs;
        __sincosf(ang, &sn, &cs);
        o1[j] = (_Float16)(k1v[j] * cs - k2v[j] * sn);
        o2[j] = (_Float16)(k2v[j] * cs + k1v[j] * sn);
      }
      *(v8h*)&ldsK[stg_key * DHEAD + stg_seg * 8]      = o1;  // 16B ds stores
      *(v8h*)&ldsK[stg_key * DHEAD + stg_seg * 8 + 64] = o2;

      const int dv0 = stg_seg * 16;
      const float vf[16] = {va0.x,va0.y,va0.z,va0.w, va1.x,va1.y,va1.z,va1.w,
                            va2.x,va2.y,va2.z,va2.w, va3.x,va3.y,va3.z,va3.w};
      #pragma unroll
      for (int j = 0; j < 16; ++j)
        ldsVT[(dv0 + j) * KTILE + stg_key] = (_Float16)vf[j];
    }
    __syncthreads();

    // ---- issue next block's global loads now; latency hides under compute -
    if (kb + 1 < nkb) {
      const int tk = (kb + 1) * KTILE + stg_key;
      const float* kp = k + ((size_t)(bt + tk) * HEADS + h) * DHEAD + stg_seg * 8;
      ka0 = *(const float4*)(kp);      ka1 = *(const float4*)(kp + 4);
      kb0 = *(const float4*)(kp + 64); kb1 = *(const float4*)(kp + 68);
      const float* vp = v + ((size_t)(bt + tk) * HEADS + h) * DVAL + stg_seg * 16;
      va0 = *(const float4*)(vp);      va1 = *(const float4*)(vp + 4);
      va2 = *(const float4*)(vp + 8);  va3 = *(const float4*)(vp + 12);
    }
    if (kb + 2 < nkb) {                 // warm L2 for the block after that
      const int tk = (kb + 2) * KTILE + stg_key;
      __builtin_prefetch(k + ((size_t)(bt + tk) * HEADS + h) * DHEAD + stg_seg * 8);
      __builtin_prefetch(v + ((size_t)(bt + tk) * HEADS + h) * DVAL + stg_seg * 16);
    }

    // ---- per-wave compute: skip blocks fully above this wave's diagonal ---
    if (kbase <= wq + 15) {
      // S = Q * K^T  (two 16x16 key subtiles, contraction 128 = 4 x 32)
      v8f sa, sb;
      #pragma unroll
      for (int e = 0; e < 8; ++e) { sa[e] = 0.0f; sb[e] = 0.0f; }
      #pragma unroll
      for (int s = 0; s < 4; ++s) {
        // B-layout: lane holds key col n = lane%16; contraction runs 8*lh..+7 and 16+8*lh..+7
        v16h bk0, bk1;
        {
          const v8h b0l = *(const v8h*)&ldsK[(ln16)      * DHEAD + 32*s + 8*lh];
          const v8h b0h = *(const v8h*)&ldsK[(ln16)      * DHEAD + 32*s + 16 + 8*lh];
          const v8h b1l = *(const v8h*)&ldsK[(16 + ln16) * DHEAD + 32*s + 8*lh];
          const v8h b1h = *(const v8h*)&ldsK[(16 + ln16) * DHEAD + 32*s + 16 + 8*lh];
          #pragma unroll
          for (int j = 0; j < 8; ++j) { bk0[j] = b0l[j]; bk0[j+8] = b0h[j]; bk1[j] = b1l[j]; bk1[j+8] = b1h[j]; }
        }
        sa = __builtin_amdgcn_wmma_f32_16x16x32_f16(false, aq[s], false, bk0,
                                                    (short)0, sa, false, false);
        sb = __builtin_amdgcn_wmma_f32_16x16x32_f16(false, aq[s], false, bk1,
                                                    (short)0, sb, false, false);
      }

      // causal mask + online softmax (C-layout: row = r + 8*lh, col = lane%16)
      const int k0 = kbase + ln16;
      const int k1i = k0 + 16;
      float pa[8], pb[8];
      #pragma unroll
      for (int r = 0; r < 8; ++r) {
        const int qr = wq + r + 8 * lh;
        const float va = (k0  <= qr) ? sa[r] : -INFINITY;
        const float vb = (k1i <= qr) ? sb[r] : -INFINITY;
        const float mx   = rowmax16(fmaxf(va, vb));
        const float mnew = fmaxf(mrow[r], mx);
        const float sc   = __expf(mrow[r] - mnew);    // 0 on first block
        const float ea   = __expf(va - mnew);
        const float eb   = __expf(vb - mnew);
        pa[r] = ea; pb[r] = eb;
        lrow[r] = lrow[r] * sc + rowsum16(ea + eb);
        mrow[r] = mnew;
        #pragma unroll
        for (int d = 0; d < 8; ++d) oacc[d][r] *= sc;
      }

      // C-layout -> A-layout for P via per-wave LDS scratch ([qrow][key])
      #pragma unroll
      for (int r = 0; r < 8; ++r) {
        ldsP[wave][(r + 8*lh) * KTILE + ln16]      = (_Float16)pa[r];
        ldsP[wave][(r + 8*lh) * KTILE + 16 + ln16] = (_Float16)pb[r];
      }
      asm volatile("s_wait_dscnt 0" ::: "memory");   // same-wave cross-lane RAW
      v16h pf;
      {
        const v8h plo = *(const v8h*)&ldsP[wave][ln16 * KTILE + 8*lh];
        const v8h phi = *(const v8h*)&ldsP[wave][ln16 * KTILE + 16 + 8*lh];
        #pragma unroll
        for (int j = 0; j < 8; ++j) { pf[j] = plo[j]; pf[j+8] = phi[j]; }
      }

      // O += P * V  (8 dv-subtiles, contraction = 32 keys, one WMMA each)
      #pragma unroll
      for (int d = 0; d < 8; ++d) {
        v16h bv;
        {
          const v8h vl = *(const v8h*)&ldsVT[(d*16 + ln16) * KTILE + 8*lh];
          const v8h vh = *(const v8h*)&ldsVT[(d*16 + ln16) * KTILE + 16 + 8*lh];
          #pragma unroll
          for (int j = 0; j < 8; ++j) { bv[j] = vl[j]; bv[j+8] = vh[j]; }
        }
        oacc[d] = __builtin_amdgcn_wmma_f32_16x16x32_f16(false, pf, false, bv,
                                                         (short)0, oacc[d], false, false);
      }
    }
    __syncthreads();
  }

  // -------- epilogue: normalize and store ----------------------------------
  #pragma unroll
  for (int r = 0; r < 8; ++r) {
    const int qr = wq + r + 8 * lh;
    const float inv = 1.0f / lrow[r];
    const size_t obase = ((size_t)(bt + qr) * HEADS + h) * DVAL;
    #pragma unroll
    for (int d = 0; d < 8; ++d)
      out[obase + d * 16 + ln16] = oacc[d][r] * inv;
  }
}

extern "C" void kernel_launch(void* const* d_in, const int* in_sizes, int n_in,
                              void* d_out, int out_size, void* d_ws, size_t ws_size,
                              hipStream_t stream) {
  const float* q = (const float*)d_in[0];
  const float* k = (const float*)d_in[1];
  const float* v = (const float*)d_in[2];
  const int* start_index = (const int*)d_in[3];
  float* out = (float*)d_out;

  dim3 grid(CHUNKL / QTILE, (BATCH * SEQ) / CHUNKL, HEADS);  // (8, 16, 16)
  dim3 block(256);
  mega_attn_kernel<<<grid, block, 0, stream>>>(q, k, v, start_index, out);
}
